// VariableSelectionNetwork_17781164605962
// MI455X (gfx1250) — matrix operand order repaired and checked by reading, not compiled, not run on hardware
//
#include <hip/hip_runtime.h>

#define BTOK 16384      // B*T
#define FDIM 32
#define HDIM 256
#define FH   8192       // F*H
#define TB   64         // tokens per block
#define SELN 288        // 256 (sw1) + 32 (ssw)
#define SELT 18         // SELN/16 N-tiles
#define LDH  264        // padded ushort row stride for LDS bf16 buffers
#define LDF  260        // padded float row stride for LDS f32 buffer
#define EPSN 1e-5f

typedef __attribute__((ext_vector_type(16))) __bf16 v16bf;
typedef __attribute__((ext_vector_type(8)))  float  v8f;

__device__ __forceinline__ unsigned short f2bf(float f) {
  union { float f; unsigned int u; } c; c.f = f;
  unsigned int u = c.u + 0x7fffu + ((c.u >> 16) & 1u);
  return (unsigned short)(u >> 16);
}
__device__ __forceinline__ float bf2f(unsigned short b) {
  union { unsigned int u; float f; } c; c.u = ((unsigned int)b) << 16;
  return c.f;
}
__device__ __forceinline__ v8f vzero() {
  v8f z = {0.f,0.f,0.f,0.f,0.f,0.f,0.f,0.f}; return z;
}
__device__ __forceinline__ v8f wmma_bf16(v16bf a, v16bf b, v8f c) {
  // D = A(16x32 bf16) x B(32x16 bf16) + C(16x16 f32)
  return __builtin_amdgcn_wmma_f32_16x16x32_bf16(false, a, false, b, (short)0, c,
                                                 false, false);
}

// A fragment 16x32 bf16 from a row-major ushort buffer (LDS or global).
// Layout: lanes 0-15 row M=lane, K 0..15; lanes 16-31 same rows,
// K 16..31 (matches swizzled B). Two contiguous 16B runs -> compiler emits 2x b128 loads.
__device__ __forceinline__ v16bf load_a(const unsigned short* __restrict__ base,
                                        size_t row0, size_t ld, int k0, int lane) {
  const unsigned short* p = base + (row0 + (size_t)(lane & 15)) * ld + k0;
  const int hi = (lane >> 4) & 1;
  union { v16bf v; unsigned short u[16]; } r;
  #pragma unroll
  for (int j = 0; j < 8; ++j) r.u[j] = p[hi * 16 + j];
  #pragma unroll
  for (int j = 0; j < 8; ++j) r.u[8 + j] = p[hi * 16 + 8 + j];
  return r.v;
}

// B fragment from pre-swizzled weights: each 32x16 (KxN) tile stored as 512
// contiguous bf16; lane l's 16 fragment elements at tile_base + l*16.
// One aligned 32-byte load per lane.
__device__ __forceinline__ v16bf load_b_swz(const unsigned short* __restrict__ base,
                                            int kt, int nTiles, int nt, int lane) {
  const unsigned short* p =
      base + ((((size_t)kt * nTiles + nt) << 5) + (size_t)lane) * 16;
  return *(const v16bf*)p;
}

// 64x256 GEMM: A (64xHDIM bf16, LDS, ld=LDH) x Bswz (HDIMxHDIM swizzled bf16).
// Wave owns M-tile mt, N-tiles (nh*8 .. nh*8+7).
__device__ __forceinline__ void gemm_64x128(const unsigned short* __restrict__ A,
                                            const unsigned short* __restrict__ B,
                                            int mt, int nh, int lane, v8f (&acc)[8]) {
  #pragma unroll
  for (int i = 0; i < 8; ++i) acc[i] = vzero();
  for (int ks = 0; ks < HDIM; ks += 32) {
    v16bf a = load_a(A, (size_t)mt * 16, LDH, ks, lane);
    const int kt = ks >> 5;
    #pragma unroll
    for (int i = 0; i < 8; ++i) {
      v16bf b = load_b_swz(B, kt, HDIM / 16, nh * 8 + i, lane);
      acc[i] = wmma_bf16(a, b, acc[i]);
    }
  }
}

// ---------------- weight converters (f32 -> swizzled bf16 tiles) -------------
// w2/wg/wf: (FDIM, 256, 256). Per f: tile = kt*16+nt (kt in 0..7, nt in 0..15),
// element (lane, j) <- in[f][kt*32 + (lane>>4)*16 + j][nt*16 + (lane&15)].
__global__ void swz_w_kernel(const float* __restrict__ in,
                             unsigned short* __restrict__ outb) {
  const size_t i = (size_t)blockIdx.x * 256 + threadIdx.x;  // FDIM*65536 total
  const int f      = (int)(i >> 16);
  const int rem    = (int)(i & 65535);
  const int tile   = rem >> 9;
  const int within = rem & 511;
  const int lane   = within >> 4;
  const int j      = within & 15;
  const int kt = tile >> 4, nt = tile & 15;
  const int k   = kt * 32 + ((lane >> 4) << 4) + j;
  const int col = nt * 16 + (lane & 15);
  outb[i] = f2bf(in[((size_t)f * HDIM + k) * HDIM + col]);
}

// [sw1 | ssw] combined (8192 x 288) -> swizzled tiles (256 kt x 18 nt).
__global__ void swz_selw_kernel(const float* __restrict__ sw1,
                                const float* __restrict__ ssw,
                                unsigned short* __restrict__ outb) {
  const size_t i = (size_t)blockIdx.x * 256 + threadIdx.x;  // FH*SELN total
  const size_t tile = i >> 9;
  const int within = (int)(i & 511);
  const int lane = within >> 4, j = within & 15;
  const int kt = (int)(tile / SELT), nt = (int)(tile % SELT);
  const int k   = kt * 32 + ((lane >> 4) << 4) + j;
  const int col = nt * 16 + (lane & 15);
  const float v = (col < HDIM) ? sw1[(size_t)k * HDIM + col]
                               : ssw[(size_t)k * FDIM + (col - HDIM)];
  outb[i] = f2bf(v);
}

// ---------------- stage A: per-feature GRN ----------------
__global__ __launch_bounds__(256) void grn_feat_kernel(
    const float* __restrict__ x,
    const float* __restrict__ w1, const float* __restrict__ b1,
    const unsigned short* __restrict__ w2b, const float* __restrict__ b2,
    const unsigned short* __restrict__ wgb, const float* __restrict__ bgv,
    const unsigned short* __restrict__ wfb, const float* __restrict__ bfv,
    const float* __restrict__ gamma, const float* __restrict__ beta,
    const float* __restrict__ wsv, const float* __restrict__ bsv,
    unsigned short* __restrict__ stk)   // (BTOK, FDIM, HDIM) bf16
{
  extern __shared__ char smem[];
  float*          xbuf = (float*)smem;                                  // 64 f32
  unsigned short* h1   = (unsigned short*)(smem + 256);                 // 64*LDH
  unsigned short* h2   = (unsigned short*)(smem + 256 + 2 * TB * LDH);  // 64*LDH
  float*          tmp  = (float*)(smem + 256 + 4 * TB * LDH);           // 64*LDF f32

  const int f    = blockIdx.y;
  const int t0   = blockIdx.x * TB;
  const int tid  = threadIdx.x;
  const int lane = tid & 31;
  const int wave = tid >> 5;
  const int fH   = f * HDIM;

  if (tid < TB) xbuf[tid] = x[(size_t)(t0 + tid) * FDIM + f];
  __syncthreads();

  // phase 1: h1 = elu(x * w1 + b1)   (one column per thread)
  {
    const float w1v = w1[fH + tid];
    const float b1v = b1[fH + tid];
    for (int m = 0; m < TB; ++m) {
      float v = xbuf[m] * w1v + b1v;
      v = v > 0.f ? v : (__expf(v) - 1.f);
      h1[m * LDH + tid] = f2bf(v);
    }
  }
  __syncthreads();

  const int mt = wave & 3;   // M-tile
  const int nh = wave >> 2;  // N half
  v8f acc[8];

  // phase 2: h2 = h1 @ w2 + b2  -> bf16 LDS
  gemm_64x128(h1, w2b + (size_t)f * HDIM * HDIM, mt, nh, lane, acc);
  #pragma unroll
  for (int i = 0; i < 8; ++i) {
    const int col = (nh * 8 + i) * 16 + (lane & 15);
    const float bias = b2[fH + col];
    #pragma unroll
    for (int r = 0; r < 8; ++r) {
      const int row = mt * 16 + ((lane >> 4) << 3) + r;
      h2[row * LDH + col] = f2bf(acc[i][r] + bias);
    }
  }
  __syncthreads();

  // phase 3a: tmp = sigmoid(h2 @ wg + bg)
  gemm_64x128(h2, wgb + (size_t)f * HDIM * HDIM, mt, nh, lane, acc);
  #pragma unroll
  for (int i = 0; i < 8; ++i) {
    const int col = (nh * 8 + i) * 16 + (lane & 15);
    const float bias = bgv[fH + col];
    #pragma unroll
    for (int r = 0; r < 8; ++r) {
      const int row = mt * 16 + ((lane >> 4) << 3) + r;
      tmp[row * LDF + col] = 1.f / (1.f + __expf(-(acc[i][r] + bias)));
    }
  }
  // phase 3b: tmp = tmp * (h2 @ wf + bf) + (x*ws + bs)   (wave-private tiles)
  gemm_64x128(h2, wfb + (size_t)f * HDIM * HDIM, mt, nh, lane, acc);
  #pragma unroll
  for (int i = 0; i < 8; ++i) {
    const int col = (nh * 8 + i) * 16 + (lane & 15);
    const float bias = bfv[fH + col];
    const float wsc = wsv[fH + col], bsc = bsv[fH + col];
    #pragma unroll
    for (int r = 0; r < 8; ++r) {
      const int row = mt * 16 + ((lane >> 4) << 3) + r;
      const float glu = tmp[row * LDF + col] * (acc[i][r] + bias);
      tmp[row * LDF + col] = glu + xbuf[row] * wsc + bsc;
    }
  }
  __syncthreads();

  // phase 4: LayerNorm over H, 8 rows per wave, bf16 out to global
  for (int rr = 0; rr < 8; ++rr) {
    const int m = wave * 8 + rr;
    float vals[8], s = 0.f, s2 = 0.f;
    #pragma unroll
    for (int j = 0; j < 8; ++j) {
      const float v = tmp[m * LDF + lane + 32 * j];
      vals[j] = v; s += v; s2 += v * v;
    }
    for (int off = 16; off > 0; off >>= 1) {
      s  += __shfl_xor(s,  off, 32);
      s2 += __shfl_xor(s2, off, 32);
    }
    const float mean = s * (1.f / HDIM);
    const float var  = s2 * (1.f / HDIM) - mean * mean;
    const float rstd = rsqrtf(var + EPSN);
    const size_t base = ((size_t)(t0 + m) * FDIM + f) * HDIM;
    #pragma unroll
    for (int j = 0; j < 8; ++j) {
      const int col = lane + 32 * j;
      const float y = (vals[j] - mean) * rstd * gamma[fH + col] + beta[fH + col];
      stk[base + col] = f2bf(y);
    }
  }
}

// ---------------- stage B: selection GRN + softmax ----------------
__global__ __launch_bounds__(256) void selection_kernel(
    const unsigned short* __restrict__ stk,   // (BTOK, FH) bf16
    const unsigned short* __restrict__ selwb, // swizzled (FH x SELN) bf16
    const float* __restrict__ sb1,
    const float* __restrict__ sw2, const float* __restrict__ sb2,
    const float* __restrict__ swg, const float* __restrict__ sbg,
    const float* __restrict__ swf, const float* __restrict__ sbf,
    const float* __restrict__ sgamma, const float* __restrict__ sbeta,
    const float* __restrict__ ssb,
    float* __restrict__ wts)                  // (BTOK, FDIM)
{
  extern __shared__ char smem[];
  float* shb = (float*)smem;                        // 64*LDF (elu(flat@sw1+sb1))
  float* srb = (float*)(smem + TB * LDF * 4);       // 64*32  (flat@ssw)

  const int t0   = blockIdx.x * TB;
  const int tid  = threadIdx.x;
  const int lane = tid & 31;
  const int wave = tid >> 5;
  const int mt   = wave & 3;
  const int ng   = wave >> 2;  // 0 -> N tiles 0..8, 1 -> N tiles 9..17

  v8f acc[9];
  #pragma unroll
  for (int i = 0; i < 9; ++i) acc[i] = vzero();

  for (int ks = 0; ks < FH; ks += 32) {
    v16bf a = load_a(stk, (size_t)(t0 + mt * 16), FH, ks, lane);
    const int kt = ks >> 5;
    #pragma unroll
    for (int i = 0; i < 9; ++i) {
      v16bf b = load_b_swz(selwb, kt, SELT, ng * 9 + i, lane);
      acc[i] = wmma_bf16(a, b, acc[i]);
    }
  }

  #pragma unroll
  for (int i = 0; i < 9; ++i) {
    const int col = (ng * 9 + i) * 16 + (lane & 15);
    #pragma unroll
    for (int r = 0; r < 8; ++r) {
      const int row = mt * 16 + ((lane >> 4) << 3) + r;
      float v = acc[i][r];
      if (col < HDIM) {
        v += sb1[col];
        v = v > 0.f ? v : (__expf(v) - 1.f);
        shb[row * LDF + col] = v;
      } else {
        srb[row * 32 + (col - HDIM)] = v;
      }
    }
  }
  __syncthreads();

  // per-token tail: 8 tokens per wave, lane = output feature j
  for (int rr = 0; rr < 8; ++rr) {
    const int m = wave * 8 + rr;
    const int j = lane;
    float s = sb2[j];
    for (int k = 0; k < HDIM; ++k) s += shb[m * LDF + k] * sw2[k * FDIM + j];
    float g = sbg[j], fv = sbf[j];
    #pragma unroll
    for (int i = 0; i < 32; ++i) {
      const float si = __shfl(s, i, 32);
      g  += si * swg[i * FDIM + j];
      fv += si * swf[i * FDIM + j];
    }
    const float sglu = fv / (1.f + __expf(-g));
    const float val = srb[m * 32 + j] + ssb[j] + sglu;
    float su = val, sq = val * val;
    for (int off = 16; off > 0; off >>= 1) {
      su += __shfl_xor(su, off, 32);
      sq += __shfl_xor(sq, off, 32);
    }
    const float mean = su * (1.f / 32.f);
    const float var  = sq * (1.f / 32.f) - mean * mean;
    const float y = (val - mean) * rsqrtf(var + EPSN) * sgamma[j] + sbeta[j];
    float mx = y;
    for (int off = 16; off > 0; off >>= 1) mx = fmaxf(mx, __shfl_xor(mx, off, 32));
    const float e = __expf(y - mx);
    float es = e;
    for (int off = 16; off > 0; off >>= 1) es += __shfl_xor(es, off, 32);
    wts[(size_t)(t0 + m) * FDIM + j] = e / es;
  }
}

// ---------------- stage C: weighted sum over features ----------------
__global__ __launch_bounds__(256) void wsum_kernel(
    const unsigned short* __restrict__ stk, const float* __restrict__ wts,
    float* __restrict__ out)
{
  const int idx = blockIdx.x * 256 + threadIdx.x;  // over BTOK*HDIM
  const int t = idx >> 8, h = idx & 255;
  const unsigned short* p = stk + (size_t)t * FH + h;
  const float* wp = wts + (size_t)t * FDIM;
  float acc = 0.f;
  #pragma unroll
  for (int f = 0; f < FDIM; ++f) acc += bf2f(p[(size_t)f * HDIM]) * wp[f];
  out[idx] = acc;
}

extern "C" void kernel_launch(void* const* d_in, const int* in_sizes, int n_in,
                              void* d_out, int out_size, void* d_ws, size_t ws_size,
                              hipStream_t stream) {
  (void)in_sizes; (void)n_in; (void)out_size; (void)ws_size;
  const float* x     = (const float*)d_in[0];
  const float* w1    = (const float*)d_in[1];
  const float* b1    = (const float*)d_in[2];
  const float* w2    = (const float*)d_in[3];
  const float* b2    = (const float*)d_in[4];
  const float* wg    = (const float*)d_in[5];
  const float* bg    = (const float*)d_in[6];
  const float* wf    = (const float*)d_in[7];
  const float* bf    = (const float*)d_in[8];
  const float* gamma = (const float*)d_in[9];
  const float* beta  = (const float*)d_in[10];
  const float* ws    = (const float*)d_in[11];
  const float* bs    = (const float*)d_in[12];
  const float* sw1   = (const float*)d_in[13];
  const float* sb1   = (const float*)d_in[14];
  const float* sw2   = (const float*)d_in[15];
  const float* sb2   = (const float*)d_in[16];
  const float* swg   = (const float*)d_in[17];
  const float* sbg   = (const float*)d_in[18];
  const float* swf   = (const float*)d_in[19];
  const float* sbf   = (const float*)d_in[20];
  const float* sgm   = (const float*)d_in[21];
  const float* sbt   = (const float*)d_in[22];
  const float* ssw   = (const float*)d_in[23];
  const float* ssb   = (const float*)d_in[24];

  // workspace layout (bytes)
  char* wsb = (char*)d_ws;
  unsigned short* stkb  = (unsigned short*)(wsb);                 // 16384*8192*2 = 256 MB
  size_t off = (size_t)BTOK * FH * 2;
  unsigned short* w2b   = (unsigned short*)(wsb + off); off += (size_t)FDIM * HDIM * HDIM * 2;
  unsigned short* wgb   = (unsigned short*)(wsb + off); off += (size_t)FDIM * HDIM * HDIM * 2;
  unsigned short* wfb   = (unsigned short*)(wsb + off); off += (size_t)FDIM * HDIM * HDIM * 2;
  unsigned short* selwb = (unsigned short*)(wsb + off); off += (size_t)FH * SELN * 2;
  float*          wts   = (float*)(wsb + off);

  const size_t nw = (size_t)FDIM * HDIM * HDIM;  // 2,097,152
  swz_w_kernel<<<(unsigned)(nw / 256), 256, 0, stream>>>(w2, w2b);
  swz_w_kernel<<<(unsigned)(nw / 256), 256, 0, stream>>>(wg, wgb);
  swz_w_kernel<<<(unsigned)(nw / 256), 256, 0, stream>>>(wf, wfb);
  const size_t ns = (size_t)FH * SELN;           // 2,359,296
  swz_selw_kernel<<<(unsigned)(ns / 256), 256, 0, stream>>>(sw1, ssw, selwb);

  const unsigned smemA = 256 + 4 * TB * LDH + 4 * TB * LDF;  // 134,400 B
  grn_feat_kernel<<<dim3(BTOK / TB, FDIM), 256, smemA, stream>>>(
      x, w1, b1, w2b, b2, wgb, bg, wfb, bf, gamma, beta, ws, bs, stkb);

  const unsigned smemB = 4 * TB * LDF + 4 * TB * 32;         // 74,752 B
  selection_kernel<<<BTOK / TB, 256, smemB, stream>>>(
      stkb, selwb, sb1, sw2, sb2, swg, sbg, swf, sbf, sgm, sbt, ssb, wts);

  wsum_kernel<<<(BTOK * HDIM) / 256, 256, 0, stream>>>(stkb, wts, (float*)d_out);
}
